// AttentionBlock_1778116461306
// MI455X (gfx1250) — compile-verified, matrix operand order checked by reading, not run on hardware
//
#include <hip/hip_runtime.h>
#include <hip/hip_bf16.h>

// Problem constants (match reference)
#define S_LEN   2048
#define D_MODEL 2048
#define N_HEAD  16
#define HEAD_DIM 128
#define D_FF    8192
#define BATCH   2
#define M_ROWS  (BATCH * S_LEN)   // 4096

typedef __attribute__((ext_vector_type(16))) __bf16 v16bf;
typedef __attribute__((ext_vector_type(8)))  __bf16 v8bf;
typedef __attribute__((ext_vector_type(2)))  __bf16 v2bf;
typedef __attribute__((ext_vector_type(8)))  float  v8f;

__device__ __forceinline__ float bf2f(__bf16 h) {
  unsigned short s = __builtin_bit_cast(unsigned short, h);
  unsigned u = ((unsigned)s) << 16;
  return __builtin_bit_cast(float, u);
}
__device__ __forceinline__ __bf16 f2bf(float f) {
  unsigned u = __builtin_bit_cast(unsigned, f);
  u = u + 0x7FFFu + ((u >> 16) & 1u);   // round-to-nearest-even
  return __builtin_bit_cast(__bf16, (unsigned short)(u >> 16));
}

__device__ __forceinline__ v8f wmma_bf16(v16bf a, v16bf b, v8f c) {
  // D = A(16x32) * B(32x16) + C(16x16), f32 accumulate
  return __builtin_amdgcn_wmma_f32_16x16x32_bf16(
      /*neg_a=*/false, a, /*neg_b=*/false, b,
      /*c_mod=*/(short)0, c, /*reuse_a=*/false, /*reuse_b=*/false);
}

// Load one 16x32 bf16 WMMA operand fragment (A layout; B is the mirrored
// read of the [N][K] operand). ISA 7.12.2: lane holds row (lane&15);
// lanes 16-31 take the +8 K-half; VGPR j<4 -> K=2j.. ; VGPR 4+j -> K=16+2j..
__device__ __forceinline__ v16bf load_frag(const __bf16* base, int rstride) {
  int lane = threadIdx.x & 31;
  const __bf16* p = base + (lane & 15) * rstride + ((lane >> 4) << 3);
  v16bf f;
#pragma unroll
  for (int j = 0; j < 4; ++j) {
    v2bf lo = *(const v2bf*)(p + 2 * j);
    v2bf hi = *(const v2bf*)(p + 16 + 2 * j);
    f[2 * j]     = lo[0]; f[2 * j + 1] = lo[1];
    f[8 + 2 * j] = hi[0]; f[9 + 2 * j] = hi[1];
  }
  return f;
}

// Async memory->LDS copy of 16 bytes (32 lanes -> 512B per instruction).
// VDST operand is the LDS byte offset (low 32 bits of the generic pointer);
// tracked with ASYNCcnt (§15.18.3, opcode 98).
__device__ __forceinline__ void async_cp16(const void* gptr, void* lptr) {
  unsigned lds_off = (unsigned)(size_t)lptr;
  asm volatile("global_load_async_to_lds_b128 %0, %1, off"
               :: "v"(lds_off), "v"(gptr) : "memory");
}
__device__ __forceinline__ void wait_async0() {
  asm volatile("s_wait_asynccnt 0" ::: "memory");
}

// ---------------------------------------------------------------------------
// f32 -> bf16 conversion (weights)
// ---------------------------------------------------------------------------
__global__ __launch_bounds__(1024) void cvt_bf16(const float* __restrict__ in,
                                                 __bf16* __restrict__ out, int n) {
  int i = blockIdx.x * 1024 + threadIdx.x;
  if (i < n) out[i] = f2bf(in[i]);
}

// ---------------------------------------------------------------------------
// Row LayerNorm: f32 [rows, D] -> bf16 [rows, D]
// ---------------------------------------------------------------------------
__global__ __launch_bounds__(256) void ln_row(const float* __restrict__ in,
                                              const float* __restrict__ g,
                                              const float* __restrict__ b,
                                              __bf16* __restrict__ out, int D) {
  __shared__ float red[256];
  const int row = blockIdx.x;
  const int tid = threadIdx.x;
  const float* p = in + (size_t)row * D;
  float vals[8];
  float s = 0.f;
#pragma unroll
  for (int e = 0; e < 8; ++e) { vals[e] = p[tid + 256 * e]; s += vals[e]; }
  red[tid] = s; __syncthreads();
  for (int st = 128; st > 0; st >>= 1) {
    if (tid < st) red[tid] += red[tid + st];
    __syncthreads();
  }
  float mu = red[0] / (float)D;
  __syncthreads();
  float q = 0.f;
#pragma unroll
  for (int e = 0; e < 8; ++e) { float d = vals[e] - mu; q += d * d; }
  red[tid] = q; __syncthreads();
  for (int st = 128; st > 0; st >>= 1) {
    if (tid < st) red[tid] += red[tid + st];
    __syncthreads();
  }
  float rstd = rsqrtf(red[0] / (float)D + 1e-5f);
#pragma unroll
  for (int e = 0; e < 8; ++e) {
    int c = tid + 256 * e;
    out[(size_t)row * D + c] = f2bf((vals[e] - mu) * rstd * g[c] + b[c]);
  }
}

// ---------------------------------------------------------------------------
// Tiled WMMA GEMM: C[M,N] = A[M,K](bf16) * W[N,K]^T(bf16) + bias.
// Block = 256 thr (8 waves as 4x2), tile 128(M) x 128(N), K step 32,
// double-buffered LDS filled by global_load_async_to_lds_b128.
// Epilogue flags: GELU, bf16 residual add, f32 out, bf16 out.
// ---------------------------------------------------------------------------
template <bool GELU, bool RES, bool OUTF, bool OUTB>
__global__ __launch_bounds__(256) void gemm_wmma(
    const __bf16* __restrict__ A, const __bf16* __restrict__ W,
    const float* __restrict__ bias, float* __restrict__ Cf,
    __bf16* __restrict__ Cb, const __bf16* __restrict__ Res,
    int M, int N, int K) {
  constexpr int LDT = 40;  // 32 + 8 pad: 80B row stride (16B aligned), no bank conflicts
  __shared__ __bf16 As[2][128 * LDT];
  __shared__ __bf16 Ws[2][128 * LDT];
  const int tid  = threadIdx.x;
  const int lane = tid & 31;
  const int wave = tid >> 5;
  const int wm = wave >> 1, wn = wave & 1;   // 4 x 2 wave grid, wave tile 32x64
  const int m0 = blockIdx.y * 128;
  const int n0 = blockIdx.x * 128;
  const int srow = tid >> 2, scol = (tid & 3) * 8;   // staging: 128 rows x 4 chunks

  v8f zero = {};
  v8f acc[2][4];
#pragma unroll
  for (int i = 0; i < 2; ++i)
#pragma unroll
    for (int j = 0; j < 4; ++j) acc[i][j] = zero;

  const __bf16* aptr = A + (size_t)(m0 + srow) * K + scol;
  const __bf16* wptr = W + (size_t)(n0 + srow) * K + scol;

  const int nk = K / 32;
  // prologue: fill buffer 0
#pragma unroll
  for (int r = 0; r < 2; ++r) {
    async_cp16(aptr + (r ? (size_t)64 * K : 0), &As[0][(srow + 64 * r) * LDT + scol]);
    async_cp16(wptr + (r ? (size_t)64 * K : 0), &Ws[0][(srow + 64 * r) * LDT + scol]);
  }

  int cur = 0;
  for (int kk = 0; kk < nk; ++kk) {
    wait_async0();
    __syncthreads();           // buffer `cur` fully resident for all waves
    if (kk + 1 < nk) {
      const int ko = (kk + 1) * 32;
#pragma unroll
      for (int r = 0; r < 2; ++r) {
        async_cp16(aptr + ko + (r ? (size_t)64 * K : 0),
                   &As[1 - cur][(srow + 64 * r) * LDT + scol]);
        async_cp16(wptr + ko + (r ? (size_t)64 * K : 0),
                   &Ws[1 - cur][(srow + 64 * r) * LDT + scol]);
      }
    }
    // compute from buffer `cur`: 6 fragment loads feed 8 WMMAs
    v16bf af0 = load_frag(&As[cur][(32 * wm) * LDT], LDT);
    v16bf af1 = load_frag(&As[cur][(32 * wm + 16) * LDT], LDT);
    v16bf bf0 = load_frag(&Ws[cur][(64 * wn) * LDT], LDT);
    v16bf bf1 = load_frag(&Ws[cur][(64 * wn + 16) * LDT], LDT);
    v16bf bf2 = load_frag(&Ws[cur][(64 * wn + 32) * LDT], LDT);
    v16bf bf3 = load_frag(&Ws[cur][(64 * wn + 48) * LDT], LDT);
    acc[0][0] = wmma_bf16(af0, bf0, acc[0][0]);
    acc[0][1] = wmma_bf16(af0, bf1, acc[0][1]);
    acc[0][2] = wmma_bf16(af0, bf2, acc[0][2]);
    acc[0][3] = wmma_bf16(af0, bf3, acc[0][3]);
    acc[1][0] = wmma_bf16(af1, bf0, acc[1][0]);
    acc[1][1] = wmma_bf16(af1, bf1, acc[1][1]);
    acc[1][2] = wmma_bf16(af1, bf2, acc[1][2]);
    acc[1][3] = wmma_bf16(af1, bf3, acc[1][3]);
    cur ^= 1;
  }

  // Epilogue. C layout: VGPR i = row i (lanes 0-15) / row i+8 (lanes 16-31).
  const int rb = (lane >> 4) << 3;
  const int cl = lane & 15;
#pragma unroll
  for (int im = 0; im < 2; ++im)
#pragma unroll
    for (int jn = 0; jn < 4; ++jn)
#pragma unroll
      for (int i = 0; i < 8; ++i) {
        int row = m0 + 32 * wm + 16 * im + i + rb;
        int col = n0 + 64 * wn + 16 * jn + cl;
        float v = acc[im][jn][i] + bias[col];
        if constexpr (GELU) {
          float x3 = v * v * v;
          v = 0.5f * v * (1.f + tanhf(0.7978845608028654f * (v + 0.044715f * x3)));
        }
        if constexpr (RES) v += bf2f(Res[(size_t)row * N + col]);
        if constexpr (OUTF) Cf[(size_t)row * N + col] = v;
        if constexpr (OUTB) Cb[(size_t)row * N + col] = f2bf(v);
      }
}

// ---------------------------------------------------------------------------
// Flash attention (causal). Grid: (S/128, H, B). Block 256 = 8 waves.
// Each wave: 16 query rows, Q frags in registers, K/V blocks of 32 in LDS.
// K block staged via async-to-LDS; V staged transposed through VGPRs.
// ---------------------------------------------------------------------------
__global__ __launch_bounds__(256) void flash_attn(
    const __bf16* __restrict__ Q, const __bf16* __restrict__ Km,
    const __bf16* __restrict__ Vm, __bf16* __restrict__ O) {
  constexpr float scale = 0.08838834764831845f;  // 1/sqrt(128)
  constexpr int LDK = 136;             // 272B rows: 16B-aligned, conflict-free frags
  __shared__ __bf16 Ks[32 * LDK];      // [key][hd]
  __shared__ __bf16 Vt[128 * 34];      // V transposed: [hd-col][key], padded
  __shared__ __bf16 Ps[8 * 16 * 34];   // per-wave P tile [row][key]
  const int tid  = threadIdx.x;
  const int lane = tid & 31;
  const int w    = tid >> 5;
  const int rb   = (lane >> 4) << 3;
  const int cl   = lane & 15;
  const int b = blockIdx.z, h = blockIdx.y;
  const int rowBlock = blockIdx.x * 128;
  const size_t headOff = (size_t)b * S_LEN * D_MODEL + (size_t)h * HEAD_DIM;

  // Q fragments: 16 rows x 128, 4 chunks of 32, loaded once from global
  v16bf qf[4];
#pragma unroll
  for (int c = 0; c < 4; ++c)
    qf[c] = load_frag(Q + headOff + (size_t)(rowBlock + w * 16) * D_MODEL + c * 32,
                      D_MODEL);

  v8f z = {};
  v8f o[8];
#pragma unroll
  for (int n = 0; n < 8; ++n) o[n] = z;
  float mrun[8], lrun[8];
#pragma unroll
  for (int i = 0; i < 8; ++i) { mrun[i] = -1e30f; lrun[i] = 0.f; }

  const int nkb = rowBlock / 32 + 4;   // causal: keys <= last row of block
  for (int kb = 0; kb < nkb; ++kb) {
    const int kbase = kb * 32;
    __syncthreads();                   // prior block's LDS reads done
    // Stage K (async, row-major) and V (transposed via VGPRs): 32 keys x 128
#pragma unroll
    for (int r = 0; r < 2; ++r) {
      int chunk = tid + 256 * r;
      int kr = chunk >> 4;
      int kc = (chunk & 15) * 8;
      async_cp16(Km + headOff + (size_t)(kbase + kr) * D_MODEL + kc,
                 &Ks[kr * LDK + kc]);
      v8bf vv = *(const v8bf*)(Vm + headOff + (size_t)(kbase + kr) * D_MODEL + kc);
#pragma unroll
      for (int e = 0; e < 8; ++e) Vt[(kc + e) * 34 + kr] = vv[e];
    }
    wait_async0();
    __syncthreads();

    // S = Q K^T over the 32-key block (two 16x16 score tiles)
    v8f st0 = z, st1 = z;
#pragma unroll
    for (int c = 0; c < 4; ++c) {
      v16bf kf0 = load_frag(&Ks[0 * LDK + c * 32], LDK);
      v16bf kf1 = load_frag(&Ks[16 * LDK + c * 32], LDK);
      st0 = wmma_bf16(qf[c], kf0, st0);
      st1 = wmma_bf16(qf[c], kf1, st1);
    }

    // scale + causal mask + block row-max (16-lane shfl reduce)
    float mb[8];
#pragma unroll
    for (int i = 0; i < 8; ++i) {
      int row = rowBlock + w * 16 + i + rb;
      float s0 = st0[i] * scale;
      float s1 = st1[i] * scale;
      if (kbase + cl > row)      s0 = -1e30f;
      if (kbase + 16 + cl > row) s1 = -1e30f;
      st0[i] = s0; st1[i] = s1;
      float m = fmaxf(s0, s1);
      m = fmaxf(m, __shfl_xor(m, 1));
      m = fmaxf(m, __shfl_xor(m, 2));
      m = fmaxf(m, __shfl_xor(m, 4));
      m = fmaxf(m, __shfl_xor(m, 8));
      mb[i] = m;
    }
    // online softmax update
#pragma unroll
    for (int i = 0; i < 8; ++i) {
      float mnew  = fmaxf(mrun[i], mb[i]);
      float alpha = __expf(mrun[i] - mnew);
      mrun[i] = mnew;
      float p0 = __expf(st0[i] - mnew);
      float p1 = __expf(st1[i] - mnew);
      st0[i] = p0; st1[i] = p1;
      float rs = p0 + p1;
      rs += __shfl_xor(rs, 1);
      rs += __shfl_xor(rs, 2);
      rs += __shfl_xor(rs, 4);
      rs += __shfl_xor(rs, 8);
      lrun[i] = lrun[i] * alpha + rs;
#pragma unroll
      for (int n = 0; n < 8; ++n) o[n][i] = o[n][i] * alpha;
    }

    // Re-stripe P (C-layout) into A-fragment layout via wave-private LDS
    __bf16* Pw = &Ps[w * 16 * 34];
#pragma unroll
    for (int i = 0; i < 8; ++i) {
      Pw[(i + rb) * 34 + cl]      = f2bf(st0[i]);
      Pw[(i + rb) * 34 + 16 + cl] = f2bf(st1[i]);
    }
    asm volatile("s_wait_dscnt 0" ::: "memory");
    v16bf pf = load_frag(Pw, 34);
    // O += P(16x32) @ V(32x128)
#pragma unroll
    for (int n = 0; n < 8; ++n) {
      v16bf vf = load_frag(&Vt[(n * 16) * 34], 34);
      o[n] = wmma_bf16(pf, vf, o[n]);
    }
  }

  // normalize + store bf16 output [B,S,H*HD]
#pragma unroll
  for (int i = 0; i < 8; ++i) {
    float inv = 1.f / lrun[i];
    int row = rowBlock + w * 16 + i + rb;
#pragma unroll
    for (int n = 0; n < 8; ++n)
      O[headOff + (size_t)row * D_MODEL + n * 16 + cl] = f2bf(o[n][i] * inv);
  }
}

// ---------------------------------------------------------------------------
// Host orchestration
// ---------------------------------------------------------------------------
extern "C" void kernel_launch(void* const* d_in, const int* in_sizes, int n_in,
                              void* d_out, int out_size, void* d_ws, size_t ws_size,
                              hipStream_t stream) {
  (void)in_sizes; (void)n_in; (void)out_size; (void)ws_size;
  const float* x    = (const float*)d_in[0];
  const float* Wq   = (const float*)d_in[1];
  const float* bq   = (const float*)d_in[2];
  const float* Wk   = (const float*)d_in[3];
  const float* bk   = (const float*)d_in[4];
  const float* Wv   = (const float*)d_in[5];
  const float* bv   = (const float*)d_in[6];
  const float* Wo   = (const float*)d_in[7];
  const float* bo   = (const float*)d_in[8];
  const float* ln1g = (const float*)d_in[9];
  const float* ln1b = (const float*)d_in[10];
  const float* ln2g = (const float*)d_in[11];
  const float* ln2b = (const float*)d_in[12];
  const float* W1   = (const float*)d_in[13];
  const float* b1   = (const float*)d_in[14];
  const float* W2   = (const float*)d_in[15];
  const float* b2   = (const float*)d_in[16];

  char* ws = (char*)d_ws;
  size_t off = 0;
  auto allocB = [&](size_t n) {
    __bf16* p = (__bf16*)(ws + off);
    off += ((n * 2 + 255) & ~(size_t)255);
    return p;
  };
  auto allocF = [&](size_t n) {
    float* p = (float*)(ws + off);
    off += ((n * 4 + 255) & ~(size_t)255);
    return p;
  };

  const size_t DD = (size_t)D_MODEL * D_MODEL;
  const size_t DF = (size_t)D_FF * D_MODEL;
  const size_t MD = (size_t)M_ROWS * D_MODEL;
  const size_t MF = (size_t)M_ROWS * D_FF;

  __bf16* wqb  = allocB(DD);
  __bf16* wkb  = allocB(DD);
  __bf16* wvb  = allocB(DD);
  __bf16* wob  = allocB(DD);
  __bf16* w1b  = allocB(DF);
  __bf16* w2b  = allocB(DF);
  __bf16* hb   = allocB(MD);   // post-LN1 activations (also residual source)
  __bf16* qb   = allocB(MD);
  __bf16* kb_  = allocB(MD);
  __bf16* vb   = allocB(MD);
  __bf16* ab   = allocB(MD);   // attention output
  __bf16* h2b  = allocB(MD);   // post-LN2 activations (also residual source)
  __bf16* actb = allocB(MF);   // MLP hidden
  float*  hres = allocF(MD);   // h + attn out-proj (f32, LN2 input)

  auto cvt = [&](const float* s, __bf16* d, size_t n) {
    cvt_bf16<<<dim3((unsigned)((n + 1023) / 1024)), dim3(1024), 0, stream>>>(
        s, d, (int)n);
  };
  cvt(Wq, wqb, DD); cvt(Wk, wkb, DD); cvt(Wv, wvb, DD); cvt(Wo, wob, DD);
  cvt(W1, w1b, DF); cvt(W2, w2b, DF);

  // LN1
  ln_row<<<dim3(M_ROWS), dim3(256), 0, stream>>>(x, ln1g, ln1b, hb, D_MODEL);

  dim3 gD(D_MODEL / 128, M_ROWS / 128);
  dim3 gF(D_FF / 128,    M_ROWS / 128);
  dim3 blk(256);

  // QKV projections
  gemm_wmma<false, false, false, true><<<gD, blk, 0, stream>>>(
      hb, wqb, bq, nullptr, qb, nullptr, M_ROWS, D_MODEL, D_MODEL);
  gemm_wmma<false, false, false, true><<<gD, blk, 0, stream>>>(
      hb, wkb, bk, nullptr, kb_, nullptr, M_ROWS, D_MODEL, D_MODEL);
  gemm_wmma<false, false, false, true><<<gD, blk, 0, stream>>>(
      hb, wvb, bv, nullptr, vb, nullptr, M_ROWS, D_MODEL, D_MODEL);

  // causal flash attention
  flash_attn<<<dim3(S_LEN / 128, N_HEAD, BATCH), blk, 0, stream>>>(qb, kb_, vb, ab);

  // out-proj + residual(h) -> f32
  gemm_wmma<false, true, true, false><<<gD, blk, 0, stream>>>(
      ab, wob, bo, hres, nullptr, hb, M_ROWS, D_MODEL, D_MODEL);

  // LN2
  ln_row<<<dim3(M_ROWS), dim3(256), 0, stream>>>(hres, ln2g, ln2b, h2b, D_MODEL);

  // MLP: GELU(h2 @ W1^T + b1) @ W2^T + b2 + h2 -> d_out (f32)
  gemm_wmma<true, false, false, true><<<gF, blk, 0, stream>>>(
      h2b, w1b, b1, nullptr, actb, nullptr, M_ROWS, D_FF, D_MODEL);
  gemm_wmma<false, true, true, false><<<gD, blk, 0, stream>>>(
      actb, w2b, b2, (float*)d_out, nullptr, h2b, M_ROWS, D_MODEL, D_FF);
}
